// DiscoMetaNet_74680891343089
// MI455X (gfx1250) — matrix operand order repaired and checked by reading, not compiled, not run on hardware
//
#include <hip/hip_runtime.h>
#include <hip/hip_bf16.h>
#include <math.h>
#include <stdint.h>

typedef __attribute__((ext_vector_type(16))) _Float16 v16h;
typedef __attribute__((ext_vector_type(8)))  _Float16 h8;
typedef __attribute__((ext_vector_type(8)))  float    v8f;
typedef __attribute__((ext_vector_type(4)))  float    f4;

#define T_    128
#define B_    256
#define A_    8
#define P_    600
#define PP_   608     // P_ padded to 32
#define HID_  256
#define MHID_ 128
#define TB_   32768   // T_*B_

__device__ __forceinline__ float sigm(float x) { return 1.0f / (1.0f + expf(-x)); }

// Merge two contiguous 8-half spans into a 16x16x32 f16 fragment.
// A: e<8 -> K=k0+8*hi+e ; e>=8 -> K=k0+16+8*hi+(e-8)  => spans at k0+8*hi and k0+16+8*hi
// B (N-major, NxK): same order as A => spans at k0+8*hi and k0+16+8*hi
__device__ __forceinline__ v16h frag(h8 lo, h8 hi) {
    return __builtin_shufflevector(lo, hi, 0,1,2,3,4,5,6,7,8,9,10,11,12,13,14,15);
}
__device__ __forceinline__ h8 zero8() {
    h8 z;
#pragma unroll
    for (int i = 0; i < 8; ++i) z[i] = (_Float16)0.0f;
    return z;
}
__device__ __forceinline__ h8 cvt8(const float* p) {          // 8 f32 -> 8 f16 (2x b128 loads)
    f4 a = *(const f4*)p, b = *(const f4*)(p + 4);
    h8 r;
#pragma unroll
    for (int i = 0; i < 4; ++i) { r[i] = (_Float16)a[i]; r[4 + i] = (_Float16)b[i]; }
    return r;
}
__device__ __forceinline__ h8 cvt8_sm(const float* p, float mx, float inv) {
    f4 a = *(const f4*)p, b = *(const f4*)(p + 4);
    h8 r;
#pragma unroll
    for (int i = 0; i < 4; ++i) {
        r[i]     = (_Float16)(expf(a[i] - mx) * inv);
        r[4 + i] = (_Float16)(expf(b[i] - mx) * inv);
    }
    return r;
}

// ===========================================================================
// Weight preparation: f32 -> f16, transposed to N-major (N x K), zero padded.
// ===========================================================================
__global__ void cvt_w1t(const float* __restrict__ W1, _Float16* __restrict__ out) {
    int idx = blockIdx.x * blockDim.x + threadIdx.x;          // 16*608
    int j = idx / PP_, k = idx - j * PP_;
    out[idx] = (k < P_) ? (_Float16)W1[k * 16 + j] : (_Float16)0.0f;
}
__global__ void cvt_headt(const float* __restrict__ W, _Float16* __restrict__ out) {
    int idx = blockIdx.x * blockDim.x + threadIdx.x;          // 608*256
    int n = idx >> 8, k = idx & 255;
    out[idx] = (n < P_) ? (_Float16)W[(size_t)k * P_ + n] : (_Float16)0.0f;
}
// traj_W (283x1024) -> (1024 x 288): K rows 0..26 = x part, 27..31 = 0, 32.. = h part
__global__ void cvt_trajt(const float* __restrict__ W, _Float16* __restrict__ out) {
    int idx = blockIdx.x * blockDim.x + threadIdx.x;          // 1024*288
    int n = idx / 288, kk = idx - n * 288;
    float v = 0.0f;
    if (kk < 27)       v = W[(size_t)kk * 1024 + n];
    else if (kk >= 32) v = W[(size_t)(kk - 5) * 1024 + n];
    out[idx] = (_Float16)v;
}
// pic_W folded xg terms -> (16 x 256)
__global__ void cvt_wct(const float* __restrict__ picW, _Float16* __restrict__ out) {
    int idx = blockIdx.x * blockDim.x + threadIdx.x;          // 16*256
    int j = idx >> 8, k = idx & 255;
    out[idx] = (_Float16)(picW[k * 16 + j] + picW[(258 + k) * 16 + j]);
}
__global__ void base_to_h16(const float* __restrict__ base, _Float16* __restrict__ b16) {
    int idx = blockIdx.x * blockDim.x + threadIdx.x;          // TB*32
    int tb = idx >> 5, k = idx & 31;
    b16[idx] = (k < 27) ? (_Float16)base[(size_t)tb * 27 + k] : (_Float16)0.0f;
}

// ===========================================================================
// _mlp2 embedding over P=600 features, compile-time softmax / dual variants.
// One wave per 16 rows; branch-free main loop; tail chunk peeled.
// ===========================================================================
template <bool SM, bool DUAL>
__global__ void emb_wmma(const float* __restrict__ src,
                         const float* __restrict__ rmax, const float* __restrict__ rsum,
                         const _Float16* __restrict__ W1t_a, const float* __restrict__ b1a,
                         const float* __restrict__ W2a, const float* __restrict__ b2a,
                         float* __restrict__ dsta, int strA, int offA,
                         const _Float16* __restrict__ W1t_b, const float* __restrict__ b1b,
                         const float* __restrict__ W2b, const float* __restrict__ b2b,
                         float* __restrict__ dstb, int strB, int offB) {
    const int lane = threadIdx.x;
    const int mt = blockIdx.x * blockDim.y + threadIdx.y;
    const int hi = lane >> 4, lm = lane & 15;
    const int m0 = mt * 16, m = m0 + lm;
    const float* srow = src + (size_t)m * P_;
    float mx = 0.0f, inv = 1.0f;
    if constexpr (SM) { mx = rmax[m]; inv = 1.0f / rsum[m]; }
    const _Float16* wa = W1t_a + (size_t)lm * PP_;
    const _Float16* wb = DUAL ? W1t_b + (size_t)lm * PP_ : nullptr;
    v8f ca = {}, cb = {};
    for (int k0 = 0; k0 < 576; k0 += 32) {                    // fully in-bounds chunks
        h8 alo, ahi;
        if constexpr (SM) {
            alo = cvt8_sm(srow + k0 + 8 * hi, mx, inv);
            ahi = cvt8_sm(srow + k0 + 16 + 8 * hi, mx, inv);
        } else {
            alo = cvt8(srow + k0 + 8 * hi);
            ahi = cvt8(srow + k0 + 16 + 8 * hi);
        }
        v16h a = frag(alo, ahi);
        v16h ba = frag(*(const h8*)(wa + k0 + 8 * hi), *(const h8*)(wa + k0 + 16 + 8 * hi));
        ca = __builtin_amdgcn_wmma_f32_16x16x32_f16(false, a, false, ba, (short)0, ca, false, false);
        if constexpr (DUAL) {
            v16h bb = frag(*(const h8*)(wb + k0 + 8 * hi), *(const h8*)(wb + k0 + 16 + 8 * hi));
            cb = __builtin_amdgcn_wmma_f32_16x16x32_f16(false, a, false, bb, (short)0, cb, false, false);
        }
    }
    {   // peeled tail chunk k0 = 576: second span OOB for hi==1 lanes -> zeros
        const int k0 = 576;
        h8 alo = SM ? cvt8_sm(srow + k0 + 8 * hi, mx, inv) : cvt8(srow + k0 + 8 * hi);
        h8 ahi = (hi == 0) ? (SM ? cvt8_sm(srow + 592, mx, inv) : cvt8(srow + 592))
                           : zero8();
        v16h a = frag(alo, ahi);
        v16h ba = frag(*(const h8*)(wa + k0 + 8 * hi), *(const h8*)(wa + k0 + 16 + 8 * hi));
        ca = __builtin_amdgcn_wmma_f32_16x16x32_f16(false, a, false, ba, (short)0, ca, false, false);
        if constexpr (DUAL) {
            v16h bb = frag(*(const h8*)(wb + k0 + 8 * hi), *(const h8*)(wb + k0 + 16 + 8 * hi));
            cb = __builtin_amdgcn_wmma_f32_16x16x32_f16(false, a, false, bb, (short)0, cb, false, false);
        }
    }
    {   // epilogue set A: relu(+b1)*W2, reduce 16 cols (lanes), +b2
        const float b1v = b1a[lm], w2v = W2a[lm], b2v = b2a[0];
#pragma unroll
        for (int r = 0; r < 8; ++r) {
            float t = fmaxf(ca[r] + b1v, 0.0f) * w2v;
#pragma unroll
            for (int o = 8; o > 0; o >>= 1) t += __shfl_xor(t, o, 16);
            if (lm == 0) dsta[(size_t)(m0 + r + 8 * hi) * strA + offA] = t + b2v;
        }
    }
    if constexpr (DUAL) {
        const float b1v = b1b[lm], w2v = W2b[lm], b2v = b2b[0];
#pragma unroll
        for (int r = 0; r < 8; ++r) {
            float t = fmaxf(cb[r] + b1v, 0.0f) * w2v;
#pragma unroll
            for (int o = 8; o > 0; o >>= 1) t += __shfl_xor(t, o, 16);
            if (lm == 0) dstb[(size_t)(m0 + r + 8 * hi) * strB + offB] = t + b2v;
        }
    }
}

// ===========================================================================
// Big GEMM: C[M,N] = Ah[M,256](f16) @ Wt(NxK f16, padded rows) + bias.
// A row-panel staged to LDS with GLOBAL_LOAD_ASYNC_TO_LDS_B128 (ASYNCcnt),
// shared by 8 waves; wave = one 16x16 tile.
// ===========================================================================
__global__ void gemm256_wmma(const _Float16* __restrict__ Ah,
                             const _Float16* __restrict__ Wt,   // (Npad x 256)
                             const float* __restrict__ bias,
                             float* __restrict__ C, int N) {
    __shared__ _Float16 Ap[16 * 256];
    const int lane = threadIdx.x, ty = threadIdx.y;
    const int tid = ty * 32 + lane;
    const int mt = blockIdx.y, m0 = mt * 16;
    // async stage A panel: 16 rows x 256 halves = 512 spans of 16B, 2 per thread
#pragma unroll
    for (int s = tid; s < 512; s += 256) {
        int row = s >> 5, kk = (s & 31) * 8;
        uint32_t lds = (uint32_t)(uintptr_t)&Ap[row * 256 + kk];
        uint64_t ga  = (uint64_t)(uintptr_t)&Ah[(size_t)(m0 + row) * 256 + kk];
        asm volatile("global_load_async_to_lds_b128 %0, %1, off"
                     :: "v"(lds), "v"(ga) : "memory");
    }
    asm volatile("s_wait_asynccnt 0" ::: "memory");
    __syncthreads();
    const int ntile = blockIdx.x * blockDim.y + ty;
    const int ntiles = (N + 15) >> 4;
    if (ntile >= ntiles) return;
    const int hi = lane >> 4, lm = lane & 15;
    const int n = ntile * 16 + lm;
    const _Float16* wp = Wt + (size_t)n * 256;
    __builtin_prefetch(wp, 0, 1);
    v8f c = {};
#pragma unroll
    for (int k0 = 0; k0 < 256; k0 += 32) {
        v16h a = frag(*(const h8*)&Ap[lm * 256 + k0 + 8 * hi],
                      *(const h8*)&Ap[lm * 256 + k0 + 16 + 8 * hi]);
        v16h b = frag(*(const h8*)(wp + k0 + 8 * hi), *(const h8*)(wp + k0 + 16 + 8 * hi));
        c = __builtin_amdgcn_wmma_f32_16x16x32_f16(false, a, false, b, (short)0, c, false, false);
    }
    if (n < N) {
        float bv = bias ? bias[n] : 0.0f;
#pragma unroll
        for (int r = 0; r < 8; ++r)
            C[(size_t)(m0 + r + 8 * hi) * N + n] = c[r] + bv;
    }
}

// N=16 variant (pic s-term): wave per m-tile, B from (16x256) f16.
__global__ void gemm256_n16(const _Float16* __restrict__ Ah,
                            const _Float16* __restrict__ Wt,    // (16 x 256)
                            float* __restrict__ C) {
    const int lane = threadIdx.x;
    const int mt = blockIdx.x * blockDim.y + threadIdx.y;
    const int hi = lane >> 4, lm = lane & 15;
    const int m0 = mt * 16;
    const _Float16* arow = Ah + (size_t)(m0 + lm) * 256;
    const _Float16* wp = Wt + (size_t)lm * 256;
    v8f c = {};
#pragma unroll
    for (int k0 = 0; k0 < 256; k0 += 32) {
        v16h a = frag(*(const h8*)(arow + k0 + 8 * hi),
                      *(const h8*)(arow + k0 + 16 + 8 * hi));
        v16h b = frag(*(const h8*)(wp + k0 + 8 * hi), *(const h8*)(wp + k0 + 16 + 8 * hi));
        c = __builtin_amdgcn_wmma_f32_16x16x32_f16(false, a, false, b, (short)0, c, false, false);
    }
#pragma unroll
    for (int r = 0; r < 8; ++r)
        C[(size_t)(m0 + r + 8 * hi) * 16 + lm] = c[r];
}

// ===========================================================================
// LSTM gate GEMM, one step: g = [x_t | h*(1-reset)] @ traj_W + traj_b
// K = 288 (32 padded x + 256 h). Panel stage: x part via async-to-LDS (pure
// copy), h part via ds stores with fused reset select. 8 waves share panel.
// ===========================================================================
__global__ void lstm_gates_wmma(const _Float16* __restrict__ base16, // (T,B,32)
                                int t,
                                const _Float16* __restrict__ h16,    // (B,256)
                                const int* __restrict__ resets,      // (T,B)
                                const _Float16* __restrict__ trajWt, // (1024 x 288)
                                const float* __restrict__ trajb,
                                float* __restrict__ g) {             // (B,1024)
    __shared__ _Float16 Ap[16 * 288];
    const int lane = threadIdx.x, ty = threadIdx.y;
    const int tid = ty * 32 + lane;
    const int mt = blockIdx.x & 15, ngrp = blockIdx.x >> 4;
    const int m0 = mt * 16;
    // spans 0..63: x part (16 rows x 4 spans). spans 64..575: h part (16 x 32).
    for (int s = tid; s < 576; s += 256) {
        if (s < 64) {
            int row = s >> 2, kk = (s & 3) * 8;
            uint32_t lds = (uint32_t)(uintptr_t)&Ap[row * 288 + kk];
            uint64_t ga  = (uint64_t)(uintptr_t)&base16[((size_t)t * B_ + m0 + row) * 32 + kk];
            asm volatile("global_load_async_to_lds_b128 %0, %1, off"
                         :: "v"(lds), "v"(ga) : "memory");
        } else {
            int u = s - 64;
            int row = u >> 5, kk = (u & 31) * 8;
            h8 v = resets[(size_t)t * B_ + m0 + row]
                       ? zero8()
                       : *(const h8*)&h16[(size_t)(m0 + row) * 256 + kk];
            *(h8*)&Ap[row * 288 + 32 + kk] = v;
        }
    }
    asm volatile("s_wait_asynccnt 0" ::: "memory");
    __syncthreads();
    const int hi = lane >> 4, lm = lane & 15;
    const int ntile = ngrp * 8 + ty;
    const int n = ntile * 16 + lm;
    const _Float16* wp = trajWt + (size_t)n * 288;
    __builtin_prefetch(wp, 0, 1);
    v8f c = {};
#pragma unroll
    for (int k0 = 0; k0 < 288; k0 += 32) {
        v16h a = frag(*(const h8*)&Ap[lm * 288 + k0 + 8 * hi],
                      *(const h8*)&Ap[lm * 288 + k0 + 16 + 8 * hi]);
        v16h b = frag(*(const h8*)(wp + k0 + 8 * hi), *(const h8*)(wp + k0 + 16 + 8 * hi));
        c = __builtin_amdgcn_wmma_f32_16x16x32_f16(false, a, false, b, (short)0, c, false, false);
    }
    const float bv = trajb[n];
#pragma unroll
    for (int r = 0; r < 8; ++r)
        g[(size_t)(m0 + r + 8 * hi) * 1024 + n] = c[r] + bv;
}

// Elementwise LSTM cell update; writes f16 h + gated f16 X, f32 c.
__global__ void lstm_cell(const float* __restrict__ g,
                          const float* __restrict__ c_in,
                          const int* __restrict__ resets, int t,
                          const float* __restrict__ gatevec,
                          _Float16* __restrict__ h_out, float* __restrict__ c_out,
                          _Float16* __restrict__ Xh) {               // (T,B,256)
    int idx = blockIdx.x * blockDim.x + threadIdx.x;                 // B_*HID_
    int b = idx >> 8, j = idx & 255;
    const float* gr = g + (size_t)b * 1024;
    float gi = gr[j], gg = gr[256 + j], gf = gr[512 + j], go = gr[768 + j];
    float rs = resets[(size_t)t * B_ + b] ? 0.0f : 1.0f;
    float cp = c_in[(size_t)b * HID_ + j] * rs;
    float cn = sigm(gf + 1.0f) * cp + sigm(gi) * tanhf(gg);
    float hn = sigm(go) * tanhf(cn);
    c_out[(size_t)b * HID_ + j] = cn;
    h_out[(size_t)b * HID_ + j] = (_Float16)hn;
    Xh[((size_t)t * B_ + b) * HID_ + j] = (_Float16)(hn * gatevec[j]);
}

// gate = meta_h @ gate_W(128x256) + gate_b
__global__ void gate_kernel(const float* __restrict__ mh, const float* __restrict__ gW,
                            const float* __restrict__ gb, float* __restrict__ gate) {
    int n = threadIdx.x;
    float s = gb[n];
    for (int k = 0; k < MHID_; ++k) s += mh[k] * gW[(size_t)k * 256 + n];
    gate[n] = s;
}

// meta_input_emb: wave per row, dot(Xh[row], mih_W) + b
__global__ void rowdot256(const _Float16* __restrict__ Xh, const float* __restrict__ w,
                          const float* __restrict__ b, float* __restrict__ out) {
    int gid = blockIdx.x * blockDim.x + threadIdx.x;
    int row = gid >> 5, lane = gid & 31;
    if (row >= TB_) return;
    const _Float16* xr = Xh + (size_t)row * 256;
    float s = 0.0f;
    for (int k = lane; k < 256; k += 32) s += (float)xr[k] * w[k];
#pragma unroll
    for (int o = 16; o > 0; o >>= 1) s += __shfl_xor(s, o, 32);
    if (lane == 0) out[row] = s + b[0];
}

// Row max and sum(exp(v-max)) over y_hat rows.
__global__ void softstats(const float* __restrict__ Y,
                          float* __restrict__ rmax, float* __restrict__ rsum) {
    int gid = blockIdx.x * blockDim.x + threadIdx.x;
    int row = gid >> 5, lane = gid & 31;
    if (row >= TB_) return;
    const float* yr = Y + (size_t)row * P_;
    float mx = -3.0e38f;
    for (int k = lane; k < P_; k += 32) mx = fmaxf(mx, yr[k]);
#pragma unroll
    for (int o = 16; o > 0; o >>= 1) mx = fmaxf(mx, __shfl_xor(mx, o, 32));
    float s = 0.0f;
    for (int k = lane; k < P_; k += 32) s += expf(yr[k] - mx);
#pragma unroll
    for (int o = 16; o > 0; o >>= 1) s += __shfl_xor(s, o, 32);
    if (lane == 0) { rmax[row] = mx; rsum[row] = s; }
}

__global__ void obs_copy(const float* __restrict__ obs, float* __restrict__ base) {
    int idx = blockIdx.x * blockDim.x + threadIdx.x;
    if (idx >= TB_ * 23) return;
    int tb = idx / 23, f = idx - tb * 23;
    base[(size_t)tb * 27 + f] = obs[idx];
}

// Two chained _conv_block's on act_feats. 8 lanes per (t,b), one per a.
__global__ void pol_kernel(const float* __restrict__ act,
                           const float* __restrict__ W1, const float* __restrict__ b1,
                           const float* __restrict__ W2, const float* __restrict__ b2,
                           float* __restrict__ pol_dst,
                           float* __restrict__ pm_dst, int pm_stride, int pm_off) {
    int idx = blockIdx.x * blockDim.x + threadIdx.x;
    int tb = idx >> 3, a = idx & 7;
    if (tb >= TB_) return;
    const float* ar = act + ((size_t)tb * A_ + a) * 9;
    float in9[9], mn9[9];
#pragma unroll
    for (int f = 0; f < 9; ++f) {
        float v = ar[f];
        in9[f] = v;
        float s = v;
#pragma unroll
        for (int o = 4; o > 0; o >>= 1) s += __shfl_xor(s, o, 8);
        mn9[f] = s * 0.125f;
    }
    float h1[16], hm[16];
#pragma unroll
    for (int j = 0; j < 16; ++j) {
        float s = b1[j];
#pragma unroll
        for (int f = 0; f < 9; ++f)
            s += in9[f] * W1[f * 16 + j] + mn9[f] * W1[(9 + f) * 16 + j];
        h1[j] = fmaxf(s, 0.0f);
        float t = h1[j];
#pragma unroll
        for (int o = 4; o > 0; o >>= 1) t += __shfl_xor(t, o, 8);
        hm[j] = t * 0.125f;
    }
#pragma unroll
    for (int j = 0; j < 2; ++j) {
        float s = b2[j];
#pragma unroll
        for (int f = 0; f < 16; ++f)
            s += h1[f] * W2[f * 2 + j] + hm[f] * W2[(16 + f) * 2 + j];
        float o2 = fmaxf(s, 0.0f);
        if (pol_dst) pol_dst[((size_t)tb * A_ + a) * 2 + j] = o2;
        float pm = o2;
#pragma unroll
        for (int o = 4; o > 0; o >>= 1) pm += __shfl_xor(pm, o, 8);
        if (a == 0) pm_dst[(size_t)tb * pm_stride + pm_off + j] = pm * 0.125f;
    }
}

// pi_hat epilogue per (t,b,a)
__global__ void pic_kernel(const float* __restrict__ s_pic, const float* __restrict__ pol,
                           const float* __restrict__ base,
                           const float* __restrict__ picW, const float* __restrict__ picb,
                           const float* __restrict__ pihW, const float* __restrict__ pihb,
                           float* __restrict__ pi_out) {
    int idx = blockIdx.x * blockDim.x + threadIdx.x;
    int tb = idx >> 3, a = idx & 7;
    if (tb >= TB_) return;
    float p0 = pol[((size_t)tb * A_ + a) * 2 + 0];
    float p1 = pol[((size_t)tb * A_ + a) * 2 + 1];
    float pm0 = base[(size_t)tb * 27 + 25];
    float pm1 = base[(size_t)tb * 27 + 26];
    float acc = pihb[0];
#pragma unroll
    for (int j = 0; j < 16; ++j) {
        float gv = s_pic[(size_t)tb * 16 + j]
                 + p0 * picW[256 * 16 + j] + p1 * picW[257 * 16 + j]
                 + pm0 * picW[514 * 16 + j] + pm1 * picW[515 * 16 + j]
                 + picb[j];
        acc += fmaxf(gv, 0.0f) * pihW[j];
    }
    pi_out[(size_t)tb * A_ + a] = acc;
}

// meta_rnn_input @ mmlp_W, deterministic per-block partial sums.
__global__ void meta_mm(const float* __restrict__ obs,
                        const float* __restrict__ yemb_m, const float* __restrict__ zemb_m,
                        const float* __restrict__ pm_m, const float* __restrict__ mie,
                        const float* __restrict__ yhe,
                        const float* __restrict__ mmlpW, const float* __restrict__ mmlpb,
                        float* __restrict__ mpart) {
    __shared__ float sm[256 * 16];
    int tid = threadIdx.x;
    int tb = blockIdx.x * 256 + tid;
    float v29[29];
#pragma unroll
    for (int f = 0; f < 23; ++f) v29[f] = obs[(size_t)tb * 23 + f];
    v29[23] = yemb_m[tb];
    v29[24] = zemb_m[tb];
    v29[25] = pm_m[(size_t)tb * 2 + 0];
    v29[26] = pm_m[(size_t)tb * 2 + 1];
    v29[27] = mie[tb];
    v29[28] = yhe[tb];
#pragma unroll
    for (int j = 0; j < 16; ++j) {
        float s = mmlpb[j];
#pragma unroll
        for (int k = 0; k < 29; ++k) s += v29[k] * mmlpW[k * 16 + j];
        sm[tid * 16 + j] = s;
    }
    __syncthreads();
    for (int st = 128; st > 0; st >>= 1) {
        if (tid < st)
            for (int j = 0; j < 16; ++j)
                sm[tid * 16 + j] += sm[(tid + st) * 16 + j];
        __syncthreads();
    }
    if (tid < 16) mpart[blockIdx.x * 16 + tid] = sm[tid];
}

// Final mean -> tiny meta LSTM cell.
__global__ void meta_final(const float* __restrict__ mpart, int nblocks,
                           const float* __restrict__ mh, const float* __restrict__ mc,
                           const float* __restrict__ mcellW, const float* __restrict__ mcellb,
                           float* __restrict__ out_h, float* __restrict__ out_c) {
    __shared__ float xa[16];
    __shared__ float gsh[512];
    int tid = threadIdx.x;
    if (tid < 16) {
        float s = 0.0f;
        for (int b = 0; b < nblocks; ++b) s += mpart[b * 16 + tid];
        xa[tid] = s / (float)TB_;
    }
    __syncthreads();
    for (int n = tid; n < 512; n += blockDim.x) {
        float s = mcellb[n];
        for (int k = 0; k < 16; ++k)    s += xa[k] * mcellW[(size_t)k * 512 + n];
        for (int k = 0; k < MHID_; ++k) s += mh[k] * mcellW[(size_t)(16 + k) * 512 + n];
        gsh[n] = s;
    }
    __syncthreads();
    if (tid < MHID_) {
        float gi = gsh[tid], gg = gsh[128 + tid], gf = gsh[256 + tid], go = gsh[384 + tid];
        float cn = sigm(gf + 1.0f) * mc[tid] + sigm(gi) * tanhf(gg);
        float hn = sigm(go) * tanhf(cn);
        out_h[tid] = hn;
        out_c[tid] = cn;
    }
}

// ===========================================================================
extern "C" void kernel_launch(void* const* d_in, const int* in_sizes, int n_in,
                              void* d_out, int out_size, void* d_ws, size_t ws_size,
                              hipStream_t stream) {
    const float* y      = (const float*)d_in[0];
    const float* z      = (const float*)d_in[1];
    const float* obs    = (const float*)d_in[2];
    const float* act    = (const float*)d_in[3];
    const int*   resets = (const int*)  d_in[4];
    const float* meta_h = (const float*)d_in[5];
    const float* meta_c = (const float*)d_in[6];
    int p = 7;
    const float* o_y_W1 = (const float*)d_in[p++]; const float* o_y_b1 = (const float*)d_in[p++];
    const float* o_y_W2 = (const float*)d_in[p++]; const float* o_y_b2 = (const float*)d_in[p++];
    const float* o_z_W1 = (const float*)d_in[p++]; const float* o_z_b1 = (const float*)d_in[p++];
    const float* o_z_W2 = (const float*)d_in[p++]; const float* o_z_b2 = (const float*)d_in[p++];
    const float* o_pol_W1 = (const float*)d_in[p++]; const float* o_pol_b1 = (const float*)d_in[p++];
    const float* o_pol_W2 = (const float*)d_in[p++]; const float* o_pol_b2 = (const float*)d_in[p++];
    const float* m_y_W1 = (const float*)d_in[p++]; const float* m_y_b1 = (const float*)d_in[p++];
    const float* m_y_W2 = (const float*)d_in[p++]; const float* m_y_b2 = (const float*)d_in[p++];
    const float* m_z_W1 = (const float*)d_in[p++]; const float* m_z_b1 = (const float*)d_in[p++];
    const float* m_z_W2 = (const float*)d_in[p++]; const float* m_z_b2 = (const float*)d_in[p++];
    const float* m_pol_W1 = (const float*)d_in[p++]; const float* m_pol_b1 = (const float*)d_in[p++];
    const float* m_pol_W2 = (const float*)d_in[p++]; const float* m_pol_b2 = (const float*)d_in[p++];
    const float* trajW = (const float*)d_in[p++]; const float* trajb = (const float*)d_in[p++];
    const float* gateW = (const float*)d_in[p++]; const float* gateb = (const float*)d_in[p++];
    const float* mihW  = (const float*)d_in[p++]; const float* mihb  = (const float*)d_in[p++];
    const float* yhW   = (const float*)d_in[p++]; const float* yhb   = (const float*)d_in[p++];
    const float* zhW   = (const float*)d_in[p++]; const float* zhb   = (const float*)d_in[p++];
    const float* picW  = (const float*)d_in[p++]; const float* picb  = (const float*)d_in[p++];
    const float* pihW  = (const float*)d_in[p++]; const float* pihb  = (const float*)d_in[p++];
    const float* mmlpW = (const float*)d_in[p++]; const float* mmlpb = (const float*)d_in[p++];
    const float* mcellW = (const float*)d_in[p++]; const float* mcellb = (const float*)d_in[p++];

    float* out = (float*)d_out;
    const size_t OFF_PI  = 0;
    const size_t OFF_Y   = OFF_PI + (size_t)TB_ * A_;
    const size_t OFF_Z   = OFF_Y + (size_t)TB_ * P_;
    const size_t OFF_MIE = OFF_Z + (size_t)TB_ * P_;
    const size_t OFF_NH  = OFF_MIE + TB_;
    const size_t OFF_NC  = OFF_NH + MHID_;
    float* pi_out = out + OFF_PI;
    float* y_hat  = out + OFF_Y;
    float* z_hat  = out + OFF_Z;
    float* mie    = out + OFF_MIE;

    // ---- workspace carve (256B aligned chunks)
    char* wsb = (char*)d_ws;
    size_t off = 0;
    auto alloc = [&](size_t bytes) {
        void* ptr = wsb + off;
        off += (bytes + 255) & ~(size_t)255;
        return ptr;
    };
    float* base_o = (float*)alloc((size_t)TB_ * 27 * 4);
    float* pol_o  = (float*)alloc((size_t)TB_ * A_ * 2 * 4);
    float* pm_m   = (float*)alloc((size_t)TB_ * 2 * 4);
    float* yemb_m = (float*)alloc((size_t)TB_ * 4);
    float* zemb_m = (float*)alloc((size_t)TB_ * 4);
    float* cbuf0  = (float*)alloc((size_t)B_ * HID_ * 4);
    float* cbuf1  = (float*)alloc((size_t)B_ * HID_ * 4);
    float* gbuf   = (float*)alloc((size_t)B_ * 1024 * 4);
    float* gatev  = (float*)alloc(256 * 4);
    float* rmax   = (float*)alloc((size_t)TB_ * 4);
    float* rsum   = (float*)alloc((size_t)TB_ * 4);
    float* yhe    = (float*)alloc((size_t)TB_ * 4);
    float* s_pic  = (float*)alloc((size_t)TB_ * 16 * 4);
    float* mpart  = (float*)alloc(128 * 16 * 4);
    _Float16* base16 = (_Float16*)alloc((size_t)TB_ * 32 * 2);
    _Float16* h16_0  = (_Float16*)alloc((size_t)B_ * HID_ * 2);
    _Float16* h16_1  = (_Float16*)alloc((size_t)B_ * HID_ * 2);
    _Float16* Xh     = (_Float16*)alloc((size_t)TB_ * HID_ * 2);
    _Float16* w1t_oy = (_Float16*)alloc((size_t)16 * PP_ * 2);
    _Float16* w1t_oz = (_Float16*)alloc((size_t)16 * PP_ * 2);
    _Float16* w1t_my = (_Float16*)alloc((size_t)16 * PP_ * 2);
    _Float16* w1t_mz = (_Float16*)alloc((size_t)16 * PP_ * 2);
    _Float16* yhWt   = (_Float16*)alloc((size_t)PP_ * 256 * 2);
    _Float16* zhWt   = (_Float16*)alloc((size_t)PP_ * 256 * 2);
    _Float16* trajWt = (_Float16*)alloc((size_t)1024 * 288 * 2);
    _Float16* Wct    = (_Float16*)alloc((size_t)16 * 256 * 2);
    _Float16* hb[2] = {h16_0, h16_1};
    float*    cb[2] = {cbuf0, cbuf1};

    // ---- stage 0: init, weight prep, input construction
    hipMemsetAsync(h16_0, 0, (size_t)B_ * HID_ * 2, stream);
    hipMemsetAsync(cbuf0, 0, (size_t)B_ * HID_ * 4, stream);
    cvt_w1t<<<16 * PP_ / 256, 256, 0, stream>>>(o_y_W1, w1t_oy);
    cvt_w1t<<<16 * PP_ / 256, 256, 0, stream>>>(o_z_W1, w1t_oz);
    cvt_w1t<<<16 * PP_ / 256, 256, 0, stream>>>(m_y_W1, w1t_my);
    cvt_w1t<<<16 * PP_ / 256, 256, 0, stream>>>(m_z_W1, w1t_mz);
    cvt_headt<<<PP_ * 256 / 256, 256, 0, stream>>>(yhW, yhWt);
    cvt_headt<<<PP_ * 256 / 256, 256, 0, stream>>>(zhW, zhWt);
    cvt_trajt<<<1024 * 288 / 256, 256, 0, stream>>>(trajW, trajWt);
    cvt_wct<<<16, 256, 0, stream>>>(picW, Wct);
    gate_kernel<<<1, 256, 0, stream>>>(meta_h, gateW, gateb, gatev);
    obs_copy<<<(TB_ * 23 + 255) / 256, 256, 0, stream>>>(obs, base_o);
    pol_kernel<<<TB_ * A_ / 256, 256, 0, stream>>>(act, o_pol_W1, o_pol_b1, o_pol_W2, o_pol_b2,
                                                   pol_o, base_o, 27, 25);
    pol_kernel<<<TB_ * A_ / 256, 256, 0, stream>>>(act, m_pol_W1, m_pol_b1, m_pol_W2, m_pol_b2,
                                                   nullptr, pm_m, 2, 0);
    {
        dim3 blk(32, 4), grd(TB_ / 16 / 4);
        emb_wmma<false, true><<<grd, blk, 0, stream>>>(y, nullptr, nullptr,
            w1t_oy, o_y_b1, o_y_W2, o_y_b2, base_o, 27, 23,
            w1t_my, m_y_b1, m_y_W2, m_y_b2, yemb_m, 1, 0);
        emb_wmma<false, true><<<grd, blk, 0, stream>>>(z, nullptr, nullptr,
            w1t_oz, o_z_b1, o_z_W2, o_z_b2, base_o, 27, 24,
            w1t_mz, m_z_b1, m_z_W2, m_z_b2, zemb_m, 1, 0);
    }
    base_to_h16<<<TB_ * 32 / 256, 256, 0, stream>>>(base_o, base16);

    // ---- stage 1: reverse masked LSTM scan (sequential in T)
    for (int s = 0; s < T_; ++s) {
        int t = T_ - 1 - s;
        int in = s & 1, ot = in ^ 1;
        lstm_gates_wmma<<<128, dim3(32, 8), 0, stream>>>(base16, t, hb[in], resets,
                                                         trajWt, trajb, gbuf);
        lstm_cell<<<B_ * HID_ / 256, 256, 0, stream>>>(gbuf, cb[in], resets, t, gatev,
                                                       hb[ot], cb[ot], Xh);
    }

    // ---- stage 2: output heads
    gemm256_wmma<<<dim3(5, TB_ / 16), dim3(32, 8), 0, stream>>>(Xh, yhWt, yhb, y_hat, P_);
    gemm256_wmma<<<dim3(5, TB_ / 16), dim3(32, 8), 0, stream>>>(Xh, zhWt, zhb, z_hat, P_);
    gemm256_n16<<<TB_ / 16 / 4, dim3(32, 4), 0, stream>>>(Xh, Wct, s_pic);
    rowdot256<<<TB_ * 32 / 256, 256, 0, stream>>>(Xh, mihW, mihb, mie);
    pic_kernel<<<TB_ * A_ / 256, 256, 0, stream>>>(s_pic, pol_o, base_o, picW, picb,
                                                   pihW, pihb, pi_out);

    // ---- stage 3: meta path
    softstats<<<TB_ * 32 / 256, 256, 0, stream>>>(y_hat, rmax, rsum);
    emb_wmma<true, false><<<TB_ / 16 / 4, dim3(32, 4), 0, stream>>>(y_hat, rmax, rsum,
        w1t_my, m_y_b1, m_y_W2, m_y_b2, yhe, 1, 0,
        nullptr, nullptr, nullptr, nullptr, nullptr, 0, 0);
    meta_mm<<<128, 256, 0, stream>>>(obs, yemb_m, zemb_m, pm_m, mie, yhe, mmlpW, mmlpb, mpart);
    meta_final<<<1, 256, 0, stream>>>(mpart, 128, meta_h, meta_c, mcellW, mcellb,
                                      out + OFF_NH, out + OFF_NC);
    (void)in_sizes; (void)n_in; (void)out_size; (void)ws_size;
}
